// ForestRuthIntegrator_72593537237647
// MI455X (gfx1250) — compile-verified, matrix operand order checked by reading, not run on hardware
//
#include <hip/hip_runtime.h>

typedef __attribute__((ext_vector_type(16))) __bf16 v16bf;
typedef __attribute__((ext_vector_type(8)))  __bf16 v8bf;
typedef __attribute__((ext_vector_type(8)))  float  v8f;

#define BATCH 4096
#define DIM   1024
#define RANK  256
#define ROWS  16                      // batch rows per workgroup
#define TPB   256                     // 8 wave32
#define XPER  ((ROWS * DIM) / TPB)    // 64 x-elements per thread

#define DTf    0.01f
#define THETAf 1.3512071919596578f    // 1/(2 - 2^(1/3))
#define C1f    (THETAf * 0.5f)
#define C2f    ((1.0f - THETAf) * 0.5f)
#define C3f    C2f
#define C4f    C1f
#define D1f    THETAf
#define D2f    (1.0f - 2.0f * THETAf)
#define D3f    THETAf

#define WMMA_BF16(a, b, c) \
  __builtin_amdgcn_wmma_f32_16x16x32_bf16(false, (a), false, (b), (short)0, (c), false, false)

// Fragment load from row-major bf16 storage (LDS or global); p points at
// element [row][kbase + (lane>>4)*8]; both halves are 16B-aligned b128 loads.
__device__ inline v16bf load_frag16(const __bf16* p) {
  v8bf lo = *(const v8bf*)(p);
  v8bf hi = *(const v8bf*)(p + 16);
  v16bf r;
#pragma unroll
  for (int i = 0; i < 8; ++i) { r[i] = lo[i]; r[8 + i] = hi[i]; }
  return r;
}

// Opaque zero: laundered through an empty asm each substep so addresses that
// add it cannot be proven loop-invariant (blocks LICM/scratch-caching of the
// B-panel load stream across the time loop), while the base pointer keeps its
// global address space (loads stay global_load_b128, not flat, so DScnt is
// reserved for the real LDS traffic).
__device__ inline size_t opaque_zero() {
  size_t z = 0;
  asm volatile("" : "+s"(z));
  return z;
}

// Pre-transpose + downconvert the constant factors once:
//   Ut[r][d] = bf16(U[d][r])   (RANK x DIM)  -> B operand of GEMM1 (h = v @ U)
//   Wt[d][r] = bf16(W[r][d])   (DIM x RANK)  -> B operand of GEMM2 (a = hh @ W)
__global__ __launch_bounds__(256) void fr_prep_kernel(
    const float* __restrict__ U, const float* __restrict__ W,
    __bf16* __restrict__ Ut, __bf16* __restrict__ Wt) {
  int idx = blockIdx.x * 256 + threadIdx.x;      // 0 .. DIM*RANK-1
  int r = idx / DIM, d = idx % DIM;
  Ut[idx] = (__bf16)U[(size_t)d * RANK + r];
  int d2 = idx / RANK, r2 = idx % RANK;
  Wt[idx] = (__bf16)W[(size_t)r2 * DIM + d2];
}

// Pair-tile GEMM panel: two 16x16 C tiles = A(16xK) x B rows [n0..n0+31],
// software-pipelined; A from LDS, B from global.
__device__ inline void gemm_pair(const __bf16* __restrict__ ap,
                                 const __bf16* __restrict__ bp0,
                                 int K, v8f& out0, v8f& out1) {
  const __bf16* bp1 = bp0 + (size_t)16 * K;
  v8f acc0 = {}, acc1 = {};
  v16bf a  = load_frag16(ap);
  v16bf b0 = load_frag16(bp0);
  v16bf b1 = load_frag16(bp1);
#pragma unroll 2
  for (int kk = 32; kk < K; kk += 32) {
    v16bf an  = load_frag16(ap + kk);
    v16bf b0n = load_frag16(bp0 + kk);
    v16bf b1n = load_frag16(bp1 + kk);
    acc0 = WMMA_BF16(a, b0, acc0);
    acc1 = WMMA_BF16(a, b1, acc1);
    a = an; b0 = b0n; b1 = b1n;
  }
  out0 = WMMA_BF16(a, b0, acc0);
  out1 = WMMA_BF16(a, b1, acc1);
}

__global__ __launch_bounds__(TPB, 4) void fr_main_kernel(
    const float* __restrict__ x_in, const float* __restrict__ v_in,
    const float* __restrict__ force, const __bf16* __restrict__ Ut,
    const __bf16* __restrict__ Wt, float* __restrict__ x_out,
    float* __restrict__ v_out, const int* __restrict__ steps_ptr) {
  __shared__ float  v_lds[ROWS * DIM];    // 64 KB: f32 v slab (precision master)
  __shared__ __bf16 vb_lds[ROWS * DIM];   // 32 KB: bf16 mirror (WMMA A source)
  __shared__ __bf16 hh_lds[ROWS * RANK];  //  8 KB: (v@U)^2 tile, bf16

  const int t    = threadIdx.x;
  const int wave = t >> 5;
  const int lane = t & 31;
  const int mrow = lane & 15;   // M row within 16-tile (C layout N = mrow too)
  const int half = lane >> 4;   // lane-half selects K-run / M+8
  const size_t base = (size_t)blockIdx.x * ROWS * DIM;

  const int steps = steps_ptr[0];

  // ---- stage slab: v -> LDS (f32 + bf16), x -> registers ----
  float x_reg[XPER];
#pragma unroll 4
  for (int i = 0; i < XPER; ++i) {
    int j = t + TPB * i;
    float vv = v_in[base + j];
    v_lds[j]  = vv;
    vb_lds[j] = (__bf16)vv;
    x_reg[i]  = x_in[base + j];
  }

  const float cs[3] = {C1f, C2f, C3f};
  const float dd[3] = {D1f, D2f, D3f};

  for (int s = 0; s < steps; ++s) {
    for (int sub = 0; sub < 3; ++sub) {
      __syncthreads();  // v_lds/vb_lds stable; hh_lds free for rewrite

      // ---- x += c*dt*v (elementwise, reads v_lds only) ----
      const float cdt = cs[sub] * DTf;
#pragma unroll 4
      for (int i = 0; i < XPER; ++i) x_reg[i] += cdt * v_lds[t + TPB * i];

      // ---- GEMM1: h = v @ U ; hh = h*h -> LDS (2 N-tiles per wave) ----
      {
        const __bf16* Ut_o = Ut + opaque_zero();  // per-substep: no LICM
        v8f acc0, acc1;
        gemm_pair(&vb_lds[mrow * DIM + half * 8],
                  &Ut_o[(size_t)(wave * 32 + mrow) * DIM + half * 8],
                  DIM, acc0, acc1);
#pragma unroll
        for (int g = 0; g < 8; ++g) {  // C layout: M = g + 8*half, N = nbase+mrow
          const int m = g + 8 * half;
          float h0 = acc0[g], h1 = acc1[g];
          hh_lds[m * RANK + wave * 32 + mrow]      = (__bf16)(h0 * h0);
          hh_lds[m * RANK + wave * 32 + 16 + mrow] = (__bf16)(h1 * h1);
        }
      }

      __syncthreads();  // hh complete; v_lds reads done -> safe to update v

      // ---- GEMM2: a = -(hh @ W) + force ; v += d*dt*a ----
      // One wave owns 8 N-tiles, processed as 4 pairs sharing the A-frag.
      const float ddt = dd[sub] * DTf;
      const __bf16* Wt_o = Wt + opaque_zero();    // per-substep: no LICM
#pragma unroll
      for (int q = 0; q < 4; ++q) {
        const int nbase = wave * 128 + q * 32;
        v8f acc0, acc1;
        gemm_pair(&hh_lds[mrow * RANK + half * 8],
                  &Wt_o[(size_t)(nbase + mrow) * RANK + half * 8],
                  RANK, acc0, acc1);
#pragma unroll
        for (int g = 0; g < 8; ++g) {
          const int m = g + 8 * half;
#pragma unroll
          for (int j = 0; j < 2; ++j) {
            const int n = nbase + j * 16 + mrow;
            float av = (j == 0) ? acc0[g] : acc1[g];
            float f  = force[base + (size_t)m * DIM + n];
            float vn = v_lds[m * DIM + n] + ddt * (f - av);
            v_lds[m * DIM + n]  = vn;
            vb_lds[m * DIM + n] = (__bf16)vn;
          }
        }
      }
    }

    __syncthreads();  // v updated by all waves before the C4 drift
#pragma unroll 4
    for (int i = 0; i < XPER; ++i) x_reg[i] += (C4f * DTf) * v_lds[t + TPB * i];
  }

  __syncthreads();
#pragma unroll 4
  for (int i = 0; i < XPER; ++i) {
    int j = t + TPB * i;
    x_out[base + j] = x_reg[i];
    v_out[base + j] = v_lds[j];
  }
}

extern "C" void kernel_launch(void* const* d_in, const int* in_sizes, int n_in,
                              void* d_out, int out_size, void* d_ws, size_t ws_size,
                              hipStream_t stream) {
  const float* x     = (const float*)d_in[0];
  const float* v     = (const float*)d_in[1];
  const float* force = (const float*)d_in[2];
  const float* U     = (const float*)d_in[3];
  const float* W     = (const float*)d_in[4];
  const int*   steps = (const int*)d_in[5];

  float* x_out = (float*)d_out;
  float* v_out = x_out + (size_t)BATCH * DIM;

  __bf16* Ut = (__bf16*)d_ws;                    // RANK x DIM bf16 (512 KB)
  __bf16* Wt = Ut + (size_t)DIM * RANK;          // DIM x RANK bf16 (512 KB)

  fr_prep_kernel<<<(DIM * RANK) / 256, 256, 0, stream>>>(U, W, Ut, Wt);
  fr_main_kernel<<<BATCH / ROWS, TPB, 0, stream>>>(x, v, force, Ut, Wt,
                                                   x_out, v_out, steps);
}
